// VAE_43868795961551
// MI455X (gfx1250) — matrix operand order repaired and checked by reading, not compiled, not run on hardware
//
#include <hip/hip_runtime.h>
#include <hip/hip_bf16.h>

typedef __attribute__((ext_vector_type(16))) _Float16 v16h;
typedef __attribute__((ext_vector_type(8)))  float    v8f;
typedef __attribute__((ext_vector_type(8)))  _Float16 h8;

// ---------------------------------------------------------------- convert
__global__ void cvt_f32_f16(const float* __restrict__ src,
                            _Float16* __restrict__ dst, int n) {
  int i = blockIdx.x * blockDim.x + threadIdx.x;
  if (i < n) dst[i] = (_Float16)src[i];
}

// convert with row padding: dst (rows, Kd), src (rows, Ks), zero tail
__global__ void cvt_pad_f32_f16(const float* __restrict__ src,
                                _Float16* __restrict__ dst,
                                int rows, int Ks, int Kd) {
  int i = blockIdx.x * blockDim.x + threadIdx.x;
  int total = rows * Kd;
  if (i >= total) return;
  int r = i / Kd, k = i - r * Kd;
  dst[i] = (_Float16)((k < Ks) ? src[(size_t)r * Ks + k] : 0.f);
}

// ------------------------------------------------- WMMA tile loaders (wave32)
// All K used here are multiples of 32 (padded at staging), so loads are
// unconditional 16B vector loads -> global_load_b128, no EXEC juggling.

// A 16x32 f16 tile:
//  lanes 0-15 (row M=lane):  v0..3 -> K 0..7,  v4..7 -> K 16..23
//  lanes 16-31 (row M=l-16): v0..3 -> K 8..15, v4..7 -> K 24..31
__device__ __forceinline__ v16h load_a_tile(const _Float16* __restrict__ A,
                                            int lda, int mBase, int k0) {
  int lane = threadIdx.x & 31;
  int half = lane >> 4;
  int r    = lane & 15;
  const _Float16* p = A + (size_t)(mBase + r) * (size_t)lda + k0 + half * 8;
  h8 lo = *(const h8*)(p);        // K = k0+half*8   .. +7  -> elems 0..7
  h8 hi = *(const h8*)(p + 16);   // K = k0+16+half*8 .. +7 -> elems 8..15
  v16h a;
#pragma unroll
  for (int i = 0; i < 8; ++i) { a[i] = lo[i]; a[8 + i] = hi[i]; }
  return a;
}

// B 32x16 f16 tile (B = W^T, W row-major (N,K)):
//  lane n<16 holds column N=n; same K layout as A (half -> +8 K, elems 8..15 -> +16 K)
__device__ __forceinline__ v16h load_b_tile(const _Float16* __restrict__ W,
                                            int ldw, int nBase, int k0) {
  int lane = threadIdx.x & 31;
  int half = lane >> 4;
  int n    = lane & 15;
  const _Float16* p = W + (size_t)(nBase + n) * (size_t)ldw + k0 + half * 8;
  h8 lo = *(const h8*)(p);        // K offsets half*8 .. +7
  h8 hi = *(const h8*)(p + 16);   // K offsets 16+half*8 .. +7
  v16h b;
#pragma unroll
  for (int i = 0; i < 8; ++i) { b[i] = lo[i]; b[8 + i] = hi[i]; }
  return b;
}

// --------------------------------------------------------------- WMMA GEMM
// C(M,N) = A(M,K) @ W(N,K)^T + bias ; optional f32 and/or f16 outputs,
// optional sigmoid. Strided-batch via grid.z (used for GRU directions).
template <int NT>
__global__ void gemm_wmma(const _Float16* __restrict__ A, int lda, long long aB,
                          const _Float16* __restrict__ W, int ldw, long long wB,
                          const float* __restrict__ bias, long long bB,
                          float* __restrict__ Cf, int ldc, long long cB,
                          _Float16* __restrict__ Ch, int ldch, long long chB,
                          int M, int N, int K, int act) {
  int lane = threadIdx.x & 31;
  int half = lane >> 4;
  int l16  = lane & 15;
  int nBase = blockIdx.x * (16 * NT);
  int mBase = blockIdx.y * 16;
  int z = blockIdx.z;
  A += (size_t)z * (size_t)aB;
  W += (size_t)z * (size_t)wB;

  v8f acc[NT] = {};
  for (int k0 = 0; k0 < K; k0 += 32) {
    v16h a = load_a_tile(A, lda, mBase, k0);
#pragma unroll
    for (int i = 0; i < NT; ++i) {
      v16h b = load_b_tile(W, ldw, nBase + 16 * i, k0);
      acc[i] = __builtin_amdgcn_wmma_f32_16x16x32_f16(
          false, a, false, b, (short)0, acc[i], false, false);
    }
  }

  const float* bz = bias ? bias + (size_t)z * (size_t)bB : nullptr;
  float*    cfz = Cf ? Cf + (size_t)z * (size_t)cB : nullptr;
  _Float16* chz = Ch ? Ch + (size_t)z * (size_t)chB : nullptr;
#pragma unroll
  for (int i = 0; i < NT; ++i) {
    int n = nBase + 16 * i + l16;
    float bv = bz ? bz[n] : 0.f;
#pragma unroll
    for (int v = 0; v < 8; ++v) {
      int m = mBase + half * 8 + v;          // C/D layout: M = half*8 + v
      float val = acc[i][v] + bv;
      if (act == 1) val = 1.f / (1.f + expf(-val));
      if (cfz) cfz[(size_t)m * (size_t)ldc + n] = val;
      if (chz) chz[(size_t)m * (size_t)ldch + n] = (_Float16)val;
    }
  }
}

// --------------------------------------------------- GRU gate pointwise ops
// Bidirectional: gi is (B*4, 6H) rows b*4+t ; gh/h/h16 are (2,B,*) ;
// writes y[b*yRow + t*yT + d*H + j] (f16). step in 0..3, bwd uses t=3-step.
__global__ void gru_pw_bidir(const float* __restrict__ gi,
                             const float* __restrict__ gh,
                             float* __restrict__ h, _Float16* __restrict__ h16,
                             _Float16* __restrict__ yout, int yRow, int yT,
                             int Bsz, int H, int step) {
  int idx = blockIdx.x * blockDim.x + threadIdx.x;
  int total = 2 * Bsz * H;
  if (idx >= total) return;
  int d   = idx / (Bsz * H);
  int rem = idx - d * Bsz * H;
  int b   = rem / H;
  int j   = rem - b * H;
  int t   = (d == 0) ? step : (3 - step);
  const float* g  = gi + ((size_t)(b * 4 + t) * (size_t)(6 * H) + (size_t)d * 3 * H);
  const float* gg = gh + (size_t)(d * Bsz + b) * (size_t)(3 * H);
  float ir = g[j],  iz = g[H + j],  in = g[2 * H + j];
  float hr = gg[j], hz = gg[H + j], hn = gg[2 * H + j];
  float r  = 1.f / (1.f + expf(-(ir + hr)));
  float zz = 1.f / (1.f + expf(-(iz + hz)));
  float nn = tanhf(in + r * hn);
  float hp = h[idx];
  float hnew = (1.f - zz) * nn + zz * hp;
  h[idx]   = hnew;
  h16[idx] = (_Float16)hnew;
  yout[(size_t)b * yRow + (size_t)t * yT + (size_t)d * H + j] = (_Float16)hnew;
}

// Unidirectional (rollout GRU); h updated in place, f16 mirror for GEMMs.
__global__ void gru_pw_uni(const float* __restrict__ gi,
                           const float* __restrict__ gh,
                           float* __restrict__ h, _Float16* __restrict__ h16,
                           int Bsz, int H) {
  int idx = blockIdx.x * blockDim.x + threadIdx.x;
  int total = Bsz * H;
  if (idx >= total) return;
  int b = idx / H;
  int j = idx - b * H;
  const float* g  = gi + (size_t)b * (size_t)(3 * H);
  const float* gg = gh + (size_t)b * (size_t)(3 * H);
  float ir = g[j],  iz = g[H + j],  in = g[2 * H + j];
  float hr = gg[j], hz = gg[H + j], hn = gg[2 * H + j];
  float r  = 1.f / (1.f + expf(-(ir + hr)));
  float zz = 1.f / (1.f + expf(-(iz + hz)));
  float nn = tanhf(in + r * hn);
  float hp = h[idx];
  float hnew = (1.f - zz) * nn + zz * hp;
  h[idx]   = hnew;
  h16[idx] = (_Float16)hnew;
}

// ------------------------------------- z = [eps*exp(.05*logvar), low_res(m), c]
__global__ void build_z(const float* __restrict__ eps,
                        const float* __restrict__ logvar,
                        const float* __restrict__ m,
                        const float* __restrict__ c,
                        _Float16* __restrict__ z16, int Bsz) {
  int idx = blockIdx.x * blockDim.x + threadIdx.x;
  int total = Bsz * 544;
  if (idx >= total) return;
  int b = idx / 544;
  int j = idx - b * 544;
  float val;
  if (j < 160) {
    val = eps[(size_t)b * 160 + j] * expf(0.05f * logvar[(size_t)b * 160 + j]);
  } else if (j < 352) {
    int jj = j - 160;
    int i  = jj / 12;
    int jc = jj - i * 12;
    float s = 0.f;
    for (int p = 0; p < 4; ++p)
      for (int q = 0; q < 4; ++q) {
        int r = i * 4 + p, scol = jc * 4 + q;
        int u = r * 49 + scol;      // (64,49) view of m[:, :, :784]
        int t = u / 784, w = u - t * 784;
        s += m[(size_t)b * 3200 + (size_t)t * 800 + w];
      }
    val = s;
  } else {
    val = c[(size_t)b * 192 + (j - 352)];
  }
  z16[idx] = (_Float16)val;
}

// --------------------------------------------------------------- host side
static void launch_gemm(hipStream_t s,
                        const _Float16* A, int lda, long long aB,
                        const _Float16* W, int ldw, long long wB,
                        const float* bias, long long bB,
                        float* Cf, int ldc, long long cB,
                        _Float16* Ch, int ldch, long long chB,
                        int M, int N, int K, int act, int batch) {
  dim3 blk(32, 1, 1);
  if (N % 64 == 0) {
    dim3 g(N / 64, M / 16, batch);
    gemm_wmma<4><<<g, blk, 0, s>>>(A, lda, aB, W, ldw, wB, bias, bB,
                                   Cf, ldc, cB, Ch, ldch, chB, M, N, K, act);
  } else if (N % 32 == 0) {
    dim3 g(N / 32, M / 16, batch);
    gemm_wmma<2><<<g, blk, 0, s>>>(A, lda, aB, W, ldw, wB, bias, bB,
                                   Cf, ldc, cB, Ch, ldch, chB, M, N, K, act);
  } else {
    dim3 g(N / 16, M / 16, batch);
    gemm_wmma<1><<<g, blk, 0, s>>>(A, lda, aB, W, ldw, wB, bias, bB,
                                   Cf, ldc, cB, Ch, ldch, chB, M, N, K, act);
  }
}

extern "C" void kernel_launch(void* const* d_in, const int* in_sizes, int n_in,
                              void* d_out, int out_size, void* d_ws,
                              size_t ws_size, hipStream_t stream) {
  const int B = 1024;
  const float* m_f   = (const float*)d_in[0];
  const float* c_f   = (const float*)d_in[1];
  const float* eps_f = (const float*)d_in[2];
  const float* Wih_m1 = (const float*)d_in[3];
  const float* Whh_m1 = (const float*)d_in[4];
  const float* bih_m1 = (const float*)d_in[5];
  const float* bhh_m1 = (const float*)d_in[6];
  const float* Wih_m2 = (const float*)d_in[7];
  const float* Whh_m2 = (const float*)d_in[8];
  const float* bih_m2 = (const float*)d_in[9];
  const float* bhh_m2 = (const float*)d_in[10];
  const float* Wih_c1 = (const float*)d_in[11];
  const float* Whh_c1 = (const float*)d_in[12];
  const float* bih_c1 = (const float*)d_in[13];
  const float* bhh_c1 = (const float*)d_in[14];
  const float* Wih_c2 = (const float*)d_in[15];
  const float* Whh_c2 = (const float*)d_in[16];
  const float* bih_c2 = (const float*)d_in[17];
  const float* bhh_c2 = (const float*)d_in[18];
  const float* Wih_g  = (const float*)d_in[19];   // melody rollout GRU
  const float* Whh_g  = (const float*)d_in[20];
  const float* bih_g  = (const float*)d_in[21];
  const float* bhh_g  = (const float*)d_in[22];
  const float* Wih_h  = (const float*)d_in[23];   // chord rollout GRU
  const float* Whh_h  = (const float*)d_in[24];
  const float* bih_h  = (const float*)d_in[25];
  const float* bhh_h  = (const float*)d_in[26];
  const float* Wmu_f  = (const float*)d_in[27];
  const float* bmu_f  = (const float*)d_in[28];
  const float* Wvar_f = (const float*)d_in[29];
  const float* bvar_f = (const float*)d_in[30];
  const float* Wlatm_f = (const float*)d_in[31];
  const float* blatm_f = (const float*)d_in[32];
  const float* Wlatc_f = (const float*)d_in[33];
  const float* blatc_f = (const float*)d_in[34];
  const float* Woutm_f = (const float*)d_in[35];
  const float* boutm_f = (const float*)d_in[36];
  const float* Woutc_f = (const float*)d_in[37];
  const float* boutc_f = (const float*)d_in[38];

  float* out = (float*)d_out;
  const size_t MEL = (size_t)B * 4 * 800;
  const size_t CHD = (size_t)B * 4 * 48;
  float* out_mel = out;
  float* out_chd = out + MEL;
  float* out_mu  = out + MEL + CHD;
  float* out_lv  = out + MEL + CHD + (size_t)B * 160;

  // bump allocator over d_ws
  size_t off = 0;
  auto alloc = [&](size_t bytes) -> void* {
    off = (off + 255) & ~(size_t)255;
    void* p = (char*)d_ws + off;
    off += bytes;
    return p;
  };
  auto aH = [&](size_t n) { return (_Float16*)alloc(n * sizeof(_Float16)); };
  auto aF = [&](size_t n) { return (float*)alloc(n * sizeof(float)); };
  auto cvt = [&](const float* s, _Float16* d, size_t n) {
    cvt_f32_f16<<<(int)((n + 255) / 256), 256, 0, stream>>>(s, d, (int)n);
  };

  // ---------- f16 copies of activations + weights ----------
  _Float16* m16 = aH(MEL);
  _Float16* c16 = aH((size_t)4096 * 64);          // K padded 48 -> 64
  _Float16* wih_m1 = aH(768 * 800);
  _Float16* whh_m1 = aH(768 * 128);
  _Float16* wih_m2 = aH(768 * 256);
  _Float16* whh_m2 = aH(768 * 128);
  _Float16* wih_c1 = aH(192 * 64);                // K padded 48 -> 64
  _Float16* whh_c1 = aH(192 * 32);
  _Float16* wih_c2 = aH(192 * 64);
  _Float16* whh_c2 = aH(192 * 32);
  _Float16* wih_g = aH((size_t)3072 * 1024);
  _Float16* whh_g = aH((size_t)3072 * 1024);
  _Float16* wih_h = aH(768 * 256);
  _Float16* whh_h = aH(768 * 256);
  _Float16* wmu   = aH(160 * 1280);
  _Float16* wvar  = aH(160 * 1280);
  _Float16* wlatm = aH(1024 * 544);
  _Float16* wlatc = aH(256 * 544);
  _Float16* woutm = aH(800 * 1024);
  _Float16* woutc = aH(48 * 256);

  cvt(m_f, m16, MEL);
  cvt_pad_f32_f16<<<(4096 * 64 + 255) / 256, 256, 0, stream>>>(
      c_f, c16, 4096, 48, 64);
  cvt(Wih_m1, wih_m1, 768 * 800); cvt(Whh_m1, whh_m1, 768 * 128);
  cvt(Wih_m2, wih_m2, 768 * 256); cvt(Whh_m2, whh_m2, 768 * 128);
  cvt_pad_f32_f16<<<(192 * 64 + 255) / 256, 256, 0, stream>>>(
      Wih_c1, wih_c1, 192, 48, 64);
  cvt(Whh_c1, whh_c1, 192 * 32);
  cvt(Wih_c2, wih_c2, 192 * 64);  cvt(Whh_c2, whh_c2, 192 * 32);
  cvt(Wih_g, wih_g, (size_t)3072 * 1024);
  cvt(Whh_g, whh_g, (size_t)3072 * 1024);
  cvt(Wih_h, wih_h, 768 * 256);   cvt(Whh_h, whh_h, 768 * 256);
  cvt(Wmu_f, wmu, 160 * 1280);    cvt(Wvar_f, wvar, 160 * 1280);
  cvt(Wlatm_f, wlatm, 1024 * 544); cvt(Wlatc_f, wlatc, 256 * 544);
  cvt(Woutm_f, woutm, 800 * 1024); cvt(Woutc_f, woutc, 48 * 256);

  // ---------- activation scratch ----------
  float*    gi_m  = aF((size_t)4096 * 768);
  float*    gh_m  = aF((size_t)2 * 1024 * 384);
  float*    h_m   = aF((size_t)2 * 1024 * 128);
  _Float16* h16_m = aH((size_t)2 * 1024 * 128);
  _Float16* y1_m  = aH((size_t)4096 * 256);
  float*    gi_c  = aF((size_t)4096 * 192);
  float*    gh_c  = aF((size_t)2 * 1024 * 96);
  float*    h_c   = aF((size_t)2 * 1024 * 32);
  _Float16* h16_c = aH((size_t)2 * 1024 * 32);
  _Float16* y1_c  = aH((size_t)4096 * 64);
  _Float16* hcat  = aH((size_t)1024 * 1280);
  _Float16* z16   = aH((size_t)1024 * 544);
  _Float16* latm  = aH((size_t)1024 * 1024);
  _Float16* latc  = aH((size_t)1024 * 256);
  float*    gi_rm = aF((size_t)1024 * 3072);
  float*    gh_rm = aF((size_t)1024 * 3072);
  float*    h_rm  = aF((size_t)1024 * 1024);
  _Float16* h16_rm = aH((size_t)1024 * 1024);
  float*    gi_rc = aF((size_t)1024 * 768);
  float*    gh_rc = aF((size_t)1024 * 768);
  float*    h_rc  = aF((size_t)1024 * 256);
  _Float16* h16_rc = aH((size_t)1024 * 256);

  // ================= m encoder, layer 1 (I=800, H=128) =================
  launch_gemm(stream, m16, 800, 0, wih_m1, 800, 0, bih_m1, 0,
              gi_m, 768, 0, nullptr, 0, 0, 4096, 768, 800, 0, 1);
  (void)hipMemsetAsync(h_m, 0, (size_t)2 * 1024 * 128 * sizeof(float), stream);
  (void)hipMemsetAsync(h16_m, 0, (size_t)2 * 1024 * 128 * sizeof(_Float16), stream);
  for (int s = 0; s < 4; ++s) {
    launch_gemm(stream, h16_m, 128, 1024 * 128, whh_m1, 128, 384 * 128,
                bhh_m1, 384, gh_m, 384, (long long)1024 * 384, nullptr, 0, 0,
                1024, 384, 128, 0, 2);
    int tot = 2 * 1024 * 128;
    gru_pw_bidir<<<(tot + 255) / 256, 256, 0, stream>>>(
        gi_m, gh_m, h_m, h16_m, y1_m, 1024, 256, 1024, 128, s);
  }
  // ================= m encoder, layer 2 (I=256, H=128) =================
  launch_gemm(stream, y1_m, 256, 0, wih_m2, 256, 0, bih_m2, 0,
              gi_m, 768, 0, nullptr, 0, 0, 4096, 768, 256, 0, 1);
  (void)hipMemsetAsync(h_m, 0, (size_t)2 * 1024 * 128 * sizeof(float), stream);
  (void)hipMemsetAsync(h16_m, 0, (size_t)2 * 1024 * 128 * sizeof(_Float16), stream);
  for (int s = 0; s < 4; ++s) {
    launch_gemm(stream, h16_m, 128, 1024 * 128, whh_m2, 128, 384 * 128,
                bhh_m2, 384, gh_m, 384, (long long)1024 * 384, nullptr, 0, 0,
                1024, 384, 128, 0, 2);
    int tot = 2 * 1024 * 128;
    gru_pw_bidir<<<(tot + 255) / 256, 256, 0, stream>>>(
        gi_m, gh_m, h_m, h16_m, hcat, 1280, 256, 1024, 128, s);
  }
  // ================= c encoder, layer 1 (I=48 padded to 64, H=32) =========
  launch_gemm(stream, c16, 64, 0, wih_c1, 64, 0, bih_c1, 0,
              gi_c, 192, 0, nullptr, 0, 0, 4096, 192, 64, 0, 1);
  (void)hipMemsetAsync(h_c, 0, (size_t)2 * 1024 * 32 * sizeof(float), stream);
  (void)hipMemsetAsync(h16_c, 0, (size_t)2 * 1024 * 32 * sizeof(_Float16), stream);
  for (int s = 0; s < 4; ++s) {
    launch_gemm(stream, h16_c, 32, 1024 * 32, whh_c1, 32, 96 * 32,
                bhh_c1, 96, gh_c, 96, (long long)1024 * 96, nullptr, 0, 0,
                1024, 96, 32, 0, 2);
    int tot = 2 * 1024 * 32;
    gru_pw_bidir<<<(tot + 255) / 256, 256, 0, stream>>>(
        gi_c, gh_c, h_c, h16_c, y1_c, 256, 64, 1024, 32, s);
  }
  // ================= c encoder, layer 2 (I=64, H=32) =================
  launch_gemm(stream, y1_c, 64, 0, wih_c2, 64, 0, bih_c2, 0,
              gi_c, 192, 0, nullptr, 0, 0, 4096, 192, 64, 0, 1);
  (void)hipMemsetAsync(h_c, 0, (size_t)2 * 1024 * 32 * sizeof(float), stream);
  (void)hipMemsetAsync(h16_c, 0, (size_t)2 * 1024 * 32 * sizeof(_Float16), stream);
  for (int s = 0; s < 4; ++s) {
    launch_gemm(stream, h16_c, 32, 1024 * 32, whh_c2, 32, 96 * 32,
                bhh_c2, 96, gh_c, 96, (long long)1024 * 96, nullptr, 0, 0,
                1024, 96, 32, 0, 2);
    int tot = 2 * 1024 * 32;
    gru_pw_bidir<<<(tot + 255) / 256, 256, 0, stream>>>(
        gi_c, gh_c, h_c, h16_c, hcat + 1024, 1280, 64, 1024, 32, s);
  }
  // ================= mu / logvar =================
  launch_gemm(stream, hcat, 1280, 0, wmu, 1280, 0, bmu_f, 0,
              out_mu, 160, 0, nullptr, 0, 0, 1024, 160, 1280, 0, 1);
  launch_gemm(stream, hcat, 1280, 0, wvar, 1280, 0, bvar_f, 0,
              out_lv, 160, 0, nullptr, 0, 0, 1024, 160, 1280, 0, 1);
  // ================= z assembly (reparam + low_res + c) =================
  build_z<<<(1024 * 544 + 255) / 256, 256, 0, stream>>>(
      eps_f, out_lv, m_f, c_f, z16, 1024);
  // ================= latent projections =================
  launch_gemm(stream, z16, 544, 0, wlatm, 544, 0, blatm_f, 0,
              nullptr, 0, 0, latm, 1024, 0, 1024, 1024, 544, 0, 1);
  launch_gemm(stream, z16, 544, 0, wlatc, 544, 0, blatc_f, 0,
              nullptr, 0, 0, latc, 256, 0, 1024, 256, 544, 0, 1);
  // ================= melody rollout (H=1024) =================
  (void)hipMemsetAsync(h_rm, 0, (size_t)1024 * 1024 * sizeof(float), stream);
  (void)hipMemsetAsync(h16_rm, 0, (size_t)1024 * 1024 * sizeof(_Float16), stream);
  for (int t = 0; t < 4; ++t) {
    const _Float16* x = (t == 0) ? latm : h16_rm;
    launch_gemm(stream, x, 1024, 0, wih_g, 1024, 0, bih_g, 0,
                gi_rm, 3072, 0, nullptr, 0, 0, 1024, 3072, 1024, 0, 1);
    launch_gemm(stream, h16_rm, 1024, 0, whh_g, 1024, 0, bhh_g, 0,
                gh_rm, 3072, 0, nullptr, 0, 0, 1024, 3072, 1024, 0, 1);
    gru_pw_uni<<<(1024 * 1024 + 255) / 256, 256, 0, stream>>>(
        gi_rm, gh_rm, h_rm, h16_rm, 1024, 1024);
    launch_gemm(stream, h16_rm, 1024, 0, woutm, 1024, 0, boutm_f, 0,
                out_mel + (size_t)t * 800, 3200, 0, nullptr, 0, 0,
                1024, 800, 1024, 1, 1);
  }
  // ================= chord rollout (H=256) =================
  (void)hipMemsetAsync(h_rc, 0, (size_t)1024 * 256 * sizeof(float), stream);
  (void)hipMemsetAsync(h16_rc, 0, (size_t)1024 * 256 * sizeof(_Float16), stream);
  for (int t = 0; t < 4; ++t) {
    const _Float16* x = (t == 0) ? latc : h16_rc;
    launch_gemm(stream, x, 256, 0, wih_h, 256, 0, bih_h, 0,
                gi_rc, 768, 0, nullptr, 0, 0, 1024, 768, 256, 0, 1);
    launch_gemm(stream, h16_rc, 256, 0, whh_h, 256, 0, bhh_h, 0,
                gh_rc, 768, 0, nullptr, 0, 0, 1024, 768, 256, 0, 1);
    gru_pw_uni<<<(1024 * 256 + 255) / 256, 256, 0, stream>>>(
        gi_rc, gh_rc, h_rc, h16_rc, 1024, 256);
    launch_gemm(stream, h16_rc, 256, 0, woutc, 256, 0, boutc_f, 0,
                out_chd + (size_t)t * 48, 192, 0, nullptr, 0, 0,
                1024, 48, 256, 1, 1);
  }
}
